// ContrastiveTokenLoss_18064632446981
// MI455X (gfx1250) — compile-verified, matrix operand order checked by reading, not run on hardware
//
#include <hip/hip_runtime.h>
#include <hip/hip_bf16.h>
#include <math.h>

// Problem constants (from reference)
#define Bb 8
#define Tt 1024
#define Dd 512
#define Kk 8192
#define Nn (Bb * Tt)          // 8192 rows
#define NNEG 16
#define INV_TEMP 10.0f        // 1 / 0.1
#define EPSN 1e-12f

#define ROWS_PER_BLOCK 64     // 4 waves x 16-row tiles
#define NWAVES 4
#define NTHREADS 128
#define NCT (Kk / 16)         // 512 column tiles
#define BTILE_BYTES (16 * Dd * 2)   // 16 KB per codebook tile

typedef __attribute__((ext_vector_type(16))) __bf16 v16bf;
typedef __attribute__((ext_vector_type(8)))  __bf16 v8bf;
typedef __attribute__((ext_vector_type(8)))  float  v8f;

#if __has_builtin(__builtin_amdgcn_tensor_load_to_lds)
#define USE_TDM 1
typedef __attribute__((ext_vector_type(4))) unsigned int v4u;
typedef __attribute__((ext_vector_type(8))) int          v8i;
typedef __attribute__((ext_vector_type(4))) int          v4i;

// Issue a TDM DMA of one 16x512 bf16 codebook tile (row-major, stride 512)
// into LDS at byte offset lds_byte_addr. D# packing per CDNA5 ISA.
__device__ __forceinline__ void tdm_load_b_tile(const __bf16* gsrc, unsigned lds_byte_addr) {
    unsigned long long ga = (unsigned long long)(size_t)gsrc;
    v4u g0 = {0u, 0u, 0u, 0u};
    g0[0] = 1u;                                              // count=1 (valid user D#)
    g0[1] = lds_byte_addr;                                   // lds_addr [63:32]
    g0[2] = (unsigned)(ga & 0xFFFFFFFFull);                  // global_addr [95:64]
    g0[3] = (unsigned)((ga >> 32) & 0x01FFFFFFull)           // global_addr [120:96]
          | (2u << 30);                                      // type=2 ("image") [127:126]
    v8i g1 = {0, 0, 0, 0, 0, 0, 0, 0};
    g1[0] = 0x10000;                     // data_size=1 -> 2 bytes (bits 17:16)
    g1[1] = (int)(512u << 16);           // tensor_dim0 = 512      (bits 79:48 lo)
    g1[2] = (int)(8192u << 16);          // tensor_dim1 = 8192     (bits 111:80 lo)
    g1[3] = (int)(512u << 16);           // tile_dim0 = 512        (bits 127:112)
    g1[4] = 16;                          // tile_dim1 = 16, tile_dim2 = 0
    g1[5] = 512;                         // tensor_dim0_stride = 512 (bits 207:160 lo)
    v4i gz = {0, 0, 0, 0};               // groups 2/3 unused for 2-D tiles
#if __clang_major__ >= 23
    v8i gz8 = {0, 0, 0, 0, 0, 0, 0, 0};
    __builtin_amdgcn_tensor_load_to_lds(g0, g1, gz, gz, gz8, 0);
#else
    __builtin_amdgcn_tensor_load_to_lds(g0, g1, gz, gz, 0);
#endif
}
#else
#define USE_TDM 0
#endif

// ---------------------------------------------------------------------------
// Kernel 0: zero the scalar output accumulator (harness poisons d_out).
// ---------------------------------------------------------------------------
__global__ void ctl_zero_kernel(float* out) {
    if (threadIdx.x == 0 && blockIdx.x == 0) out[0] = 0.0f;
}

// ---------------------------------------------------------------------------
// Kernel 1: prep. Convert x and codebook to bf16 (for the WMMA mining GEMM)
// and compute fp32 row squared-norms. One wave (32 lanes) per row.
// ---------------------------------------------------------------------------
__global__ __launch_bounds__(256) void ctl_prep_kernel(
    const float* __restrict__ x, const float* __restrict__ cbk,
    __bf16* __restrict__ xb, __bf16* __restrict__ cb,
    float* __restrict__ x2, float* __restrict__ c2)
{
    int wave = threadIdx.x >> 5;
    int lane = threadIdx.x & 31;
    int row  = blockIdx.x * 8 + wave;

    const float* src;
    __bf16* dst;
    if (row < Nn) { src = x   + (size_t)row * Dd;        dst = xb + (size_t)row * Dd; }
    else          { src = cbk + (size_t)(row - Nn) * Dd; dst = cb + (size_t)(row - Nn) * Dd; }

    float s = 0.0f;
    for (int d = lane; d < Dd; d += 32) {
        float v = src[d];
        s += v * v;
        dst[d] = (__bf16)v;
    }
    for (int off = 16; off >= 1; off >>= 1) s += __shfl_xor(s, off, 32);
    if (lane == 0) {
        if (row < Nn) x2[row] = s;
        else          c2[row - Nn] = s;
    }
}

// ---------------------------------------------------------------------------
// Kernel 2: fused cdist-GEMM (bf16 WMMA) + per-row top-16 mining + exact fp32
// InfoNCE loss. 128 blocks x 128 threads; block owns 64 rows (wave w owns
// row-tile w). Codebook tiles stream via the Tensor Data Mover into a
// double-buffered dynamic-LDS tile shared by all 4 waves.
// ---------------------------------------------------------------------------
__global__ __launch_bounds__(NTHREADS) void ctl_main_kernel(
    const float* __restrict__ x, const int* __restrict__ codes,
    const float* __restrict__ cbk,
    const __bf16* __restrict__ xb, const __bf16* __restrict__ cb,
    const float* __restrict__ x2, const float* __restrict__ c2,
    float* __restrict__ out)
{
    __shared__ __align__(16) __bf16 xs[ROWS_PER_BLOCK * Dd]; // 64 KB: 64 x-rows bf16
    __shared__ float tileBuf[NWAVES][256];                   //  4 KB: 16x16 transpose
    __shared__ float candD[NWAVES][2][16][16];               //  8 KB
    __shared__ int   candI[NWAVES][2][16][16];               //  8 KB
    __shared__ int   negIdx[ROWS_PER_BLOCK][NNEG];           //  4 KB
    __shared__ float sims[ROWS_PER_BLOCK][NNEG + 1];         //  4.25 KB
    __shared__ int   codeS[ROWS_PER_BLOCK];
    __shared__ float xs2[ROWS_PER_BLOCK];
    // B tile double buffer lives in DYNAMIC LDS (32 KB) so its byte offset is
    // known: dynamic LDS begins at __builtin_amdgcn_groupstaticsize().
    extern __shared__ __align__(16) char dynSmem[];
    __bf16* Bs = (__bf16*)dynSmem;                           // [2][16*Dd]

    const int tid  = threadIdx.x;
    const int wave = tid >> 5;
    const int lane = tid & 31;
    const int r    = lane & 15;   // row within this wave's 16-row tile
    const int h    = lane >> 4;   // half selector (K-chunk / column-half owner)
    const int gRow0 = blockIdx.x * ROWS_PER_BLOCK;
    const int myRow = wave * 16 + r;

#if USE_TDM
    const unsigned bsBase = __builtin_amdgcn_groupstaticsize();
#endif

    // ---- stage the 64 bf16 x-rows into LDS (contiguous 64 KB copy) ----
    {
        const uint4* src = (const uint4*)(xb + (size_t)gRow0 * Dd);
        uint4* dst = (uint4*)xs;
        for (int i = tid; i < (ROWS_PER_BLOCK * Dd * 2) / 16; i += NTHREADS) dst[i] = src[i];
    }
    if (tid < ROWS_PER_BLOCK) {
        codeS[tid] = codes[gRow0 + tid];
        xs2[tid]   = x2[gRow0 + tid];
    }
    // ---- preload first codebook tile (16 rows x 512, 16 KB) ----
#if USE_TDM
    if (wave == 0) {
        tdm_load_b_tile(cb, bsBase);
        __builtin_amdgcn_s_wait_tensorcnt(0);
    }
#else
    {
        const uint4* bsrc = (const uint4*)cb;
        uint4* bdst = (uint4*)Bs;
        for (int i = tid; i < 1024; i += NTHREADS) bdst[i] = bsrc[i];
    }
#endif
    __syncthreads();

    const float myX2   = xs2[myRow];
    const int   myCode = codeS[myRow];
    const __bf16* xrow = xs + (size_t)myRow * Dd;

    // register-resident sorted top-16 (ascending): td[0] smallest
    float td[16]; int ti[16];
#pragma unroll
    for (int j = 0; j < 16; ++j) { td[j] = __builtin_inff(); ti[j] = -1; }

    // ---- mining GEMM over all 512 codebook column tiles ----
    for (int ct = 0; ct < NCT; ++ct) {
        const int cur = ct & 1;

#if USE_TDM
        // DMA the next tile into the other buffer while we compute this one.
        if (wave == 0 && ct + 1 < NCT)
            tdm_load_b_tile(cb + (size_t)(ct + 1) * 16 * Dd,
                            bsBase + (unsigned)((cur ^ 1) * BTILE_BYTES));
#else
        uint4 stage[8];
        if (ct + 1 < NCT) {
            const uint4* bsrc = (const uint4*)(cb + (size_t)(ct + 1) * 16 * Dd);
#pragma unroll
            for (int i = 0; i < 8; ++i) stage[i] = bsrc[tid + i * NTHREADS];
        }
#endif

        // compute this wave's 16x16 tile: D=512 in 16 K-steps of 32
        const __bf16* brow = Bs + (size_t)cur * 16 * Dd + (size_t)r * Dd;
        v8f acc = {};
#pragma unroll 4
        for (int kk = 0; kk < Dd; kk += 32) {
            const int d0 = kk + h * 8;
            const int d1 = kk + 16 + h * 8;
            v8bf alo = *(const v8bf*)&xrow[d0];
            v8bf ahi = *(const v8bf*)&xrow[d1];
            v8bf blo = *(const v8bf*)&brow[d0];
            v8bf bhi = *(const v8bf*)&brow[d1];
            v16bf a = __builtin_shufflevector(alo, ahi, 0,1,2,3,4,5,6,7,8,9,10,11,12,13,14,15);
            v16bf b = __builtin_shufflevector(blo, bhi, 0,1,2,3,4,5,6,7,8,9,10,11,12,13,14,15);
            acc = __builtin_amdgcn_wmma_f32_16x16x32_bf16(false, a, false, b,
                                                          (short)0, acc, false, false);
        }

        // transpose through LDS: acc[v] on lane l is (m = v + 8*(l>>4), n = l&15)
#pragma unroll
        for (int v = 0; v < 8; ++v) tileBuf[wave][(v + 8 * h) * 16 + r] = acc[v];
        asm volatile("s_wait_dscnt 0" ::: "memory");

        // selection: lane (r, h) owns row r, columns h*8 .. h*8+7 of this tile
#pragma unroll
        for (int q = 0; q < 8; ++q) {
            const int n    = h * 8 + q;
            const int gcol = ct * 16 + n;
            const float dot = tileBuf[wave][r * 16 + n];
            const float d2  = fmaxf(myX2 + c2[gcol] - 2.0f * dot, 0.0f);
            if (gcol != myCode && d2 < td[15]) {
                td[15] = d2; ti[15] = gcol;
#pragma unroll
                for (int j = 15; j > 0; --j) {
                    if (td[j] < td[j - 1]) {
                        float tf = td[j]; td[j] = td[j - 1]; td[j - 1] = tf;
                        int   tx = ti[j]; ti[j] = ti[j - 1]; ti[j - 1] = tx;
                    }
                }
            }
        }

#if USE_TDM
        if (wave == 0) __builtin_amdgcn_s_wait_tensorcnt(0);
#else
        if (ct + 1 < NCT) {
            uint4* bdst = (uint4*)(Bs + (size_t)(cur ^ 1) * 16 * Dd);
#pragma unroll
            for (int i = 0; i < 8; ++i) bdst[tid + i * NTHREADS] = stage[i];
        }
#endif
        __syncthreads();
    }

    // ---- merge the two column-half top-16 lists of each wave (wave-local) ----
#pragma unroll
    for (int j = 0; j < 16; ++j) { candD[wave][h][r][j] = td[j]; candI[wave][h][r][j] = ti[j]; }
    asm volatile("s_wait_dscnt 0" ::: "memory");

    if (h == 0) {
        float fd[16]; int fi[16];
#pragma unroll
        for (int j = 0; j < 16; ++j) { fd[j] = __builtin_inff(); fi[j] = -1; }
        for (int hh = 0; hh < 2; ++hh)
            for (int j = 0; j < 16; ++j) {
                const float d2 = candD[wave][hh][r][j];
                if (d2 < fd[15]) {
                    fd[15] = d2; fi[15] = candI[wave][hh][r][j];
#pragma unroll
                    for (int j2 = 15; j2 > 0; --j2) {
                        if (fd[j2] < fd[j2 - 1]) {
                            float tf = fd[j2]; fd[j2] = fd[j2 - 1]; fd[j2 - 1] = tf;
                            int   tx = fi[j2]; fi[j2] = fi[j2 - 1]; fi[j2 - 1] = tx;
                        }
                    }
                }
            }
        for (int j = 0; j < NNEG; ++j) negIdx[myRow][j] = fi[j];
    }
    __syncthreads();

    // ---- exact fp32 cosine sims for pos + 16 negs (17 dots per row) ----
    for (int task = tid; task < ROWS_PER_BLOCK * (NNEG + 1); task += NTHREADS) {
        const int row = task / (NNEG + 1);
        const int j   = task % (NNEG + 1);
        const int cidx = (j == 0) ? codeS[row] : negIdx[row][j - 1];
        const float4* xr = (const float4*)(x   + (size_t)(gRow0 + row) * Dd);
        const float4* cr = (const float4*)(cbk + (size_t)cidx * Dd);
        float dot = 0.0f;
        for (int d = 0; d < Dd / 4; ++d) {
            float4 a = xr[d], b = cr[d];
            dot += a.x * b.x + a.y * b.y + a.z * b.z + a.w * b.w;
        }
        const float nx = fmaxf(sqrtf(xs2[row]), EPSN);
        const float nc = fmaxf(sqrtf(c2[cidx]), EPSN);
        sims[row][j] = (dot / (nx * nc)) * INV_TEMP;
    }
    __syncthreads();

    // ---- CE = logsumexp(logits) - logits[0]; accumulate mean ----
    if (tid < ROWS_PER_BLOCK) {
        float m = -__builtin_inff();
        for (int j = 0; j <= NNEG; ++j) m = fmaxf(m, sims[tid][j]);
        float s = 0.0f;
        for (int j = 0; j <= NNEG; ++j) s += __expf(sims[tid][j] - m);
        const float ce = logf(s) + m - sims[tid][0];
        atomicAdd(out, ce * (1.0f / (float)Nn));
    }
}

// ---------------------------------------------------------------------------
// Launch: inputs are (student_features f32, teacher_codes int, codebook f32).
// Workspace layout: [xb bf16 N*D][cb bf16 K*D][x2 f32 N][c2 f32 K] ~16.06 MB.
// ---------------------------------------------------------------------------
extern "C" void kernel_launch(void* const* d_in, const int* in_sizes, int n_in,
                              void* d_out, int out_size, void* d_ws, size_t ws_size,
                              hipStream_t stream) {
    const float* x     = (const float*)d_in[0];
    const int*   codes = (const int*)d_in[1];
    const float* cbk   = (const float*)d_in[2];
    float* out = (float*)d_out;

    char* ws = (char*)d_ws;
    __bf16* xb = (__bf16*)ws;
    __bf16* cb = (__bf16*)(ws + (size_t)Nn * Dd * 2);
    float*  x2 = (float*)(ws + (size_t)(Nn + Kk) * Dd * 2);
    float*  c2 = x2 + Nn;

    ctl_zero_kernel<<<1, 32, 0, stream>>>(out);
    ctl_prep_kernel<<<(Nn + Kk) / 8, 256, 0, stream>>>(x, cbk, xb, cb, x2, c2);
    ctl_main_kernel<<<Nn / ROWS_PER_BLOCK, NTHREADS, 2 * BTILE_BYTES, stream>>>(
        x, codes, cbk, xb, cb, x2, c2, out);
}
